// PatchAbstraction_15857019257405
// MI455X (gfx1250) — compile-verified, matrix operand order checked by reading, not run on hardware
//
#include <hip/hip_runtime.h>

typedef __attribute__((ext_vector_type(16))) _Float16 v16h;
typedef __attribute__((ext_vector_type(8)))  _Float16 v8h;
typedef __attribute__((ext_vector_type(8)))  float    v8f;

#define B_  8
#define N_  4096
#define C_  64
#define S_  1024
#define K_  32
#define D0_ 128
#define D1_ 256
#define EPS_ 1e-5f

// ---------------------------------------------------------------------------
// Fragment loader: 16-bit A/B fragment for v_wmma_*_16x16x32_*.
// Per ISA 7.12.2: lane L holds row (lane&15); K layout: VGPR0-3 = K[kh*8 .. kh*8+7],
// VGPR4-7 = K[16+kh*8 .. 16+kh*8+7], kh = lane>>4. Two contiguous 16B chunks.
// ---------------------------------------------------------------------------
__device__ inline v16h load_frag(const _Float16* base, int ldh) {
  const int lane = threadIdx.x & 31;
  const _Float16* p = base + (lane & 15) * ldh + (lane >> 4) * 8;
  v8h lo = *(const v8h*)(p);
  v8h hi = *(const v8h*)(p + 16);
  v16h f;
#pragma unroll
  for (int i = 0; i < 8; i++) { f[i] = lo[i]; f[i + 8] = hi[i]; }
  return f;
}

__device__ inline v8f wmma_f16(v16h a, v16h b, v8f c) {
  return __builtin_amdgcn_wmma_f32_16x16x32_f16(
      /*neg_a=*/false, a, /*neg_b=*/false, b,
      /*c_mod=*/(short)0, c, /*reuse_a=*/false, /*reuse_b=*/false);
}

// ---------------------------------------------------------------------------
// Kernel 0: fold BN into per-channel scale/bias, convert weights to f16.
// ---------------------------------------------------------------------------
__global__ void prep_kernel(
    const float* __restrict__ w0, const float* __restrict__ w1,
    const float* __restrict__ rw1, const float* __restrict__ rw2,
    const float* __restrict__ b0, const float* __restrict__ g0,
    const float* __restrict__ be0, const float* __restrict__ m0, const float* __restrict__ v0,
    const float* __restrict__ b1, const float* __restrict__ g1,
    const float* __restrict__ be1, const float* __restrict__ m1, const float* __restrict__ v1,
    const float* __restrict__ rb1, const float* __restrict__ rg1,
    const float* __restrict__ rbe1, const float* __restrict__ rm1, const float* __restrict__ rv1,
    const float* __restrict__ rb2, const float* __restrict__ rg2,
    const float* __restrict__ rbe2, const float* __restrict__ rm2, const float* __restrict__ rv2,
    _Float16* __restrict__ w0h, _Float16* __restrict__ w1h,
    _Float16* __restrict__ rw1h, _Float16* __restrict__ rw2h,
    float* __restrict__ sc0, float* __restrict__ bi0,
    float* __restrict__ sc1, float* __restrict__ bi1,
    float* __restrict__ scr1, float* __restrict__ bir1,
    float* __restrict__ scr2, float* __restrict__ bir2) {
  int i = blockIdx.x * blockDim.x + threadIdx.x;
  if (i < D0_ * 2 * C_) w0h[i] = (_Float16)w0[i];
  if (i < D1_ * D0_)    w1h[i] = (_Float16)w1[i];
  if (i < D1_ * D1_) { rw1h[i] = (_Float16)rw1[i]; rw2h[i] = (_Float16)rw2[i]; }
  if (i < D0_) {
    float a = g0[i] * rsqrtf(v0[i] + EPS_);
    sc0[i] = a; bi0[i] = (b0[i] - m0[i]) * a + be0[i];
  }
  if (i < D1_) {
    float a = g1[i] * rsqrtf(v1[i] + EPS_);
    sc1[i] = a; bi1[i] = (b1[i] - m1[i]) * a + be1[i];
    float ar1 = rg1[i] * rsqrtf(rv1[i] + EPS_);
    scr1[i] = ar1; bir1[i] = (rb1[i] - rm1[i]) * ar1 + rbe1[i];
    float ar2 = rg2[i] * rsqrtf(rv2[i] + EPS_);
    scr2[i] = ar2; bir2[i] = (rb2[i] - rm2[i]) * ar2 + rbe2[i];
  }
}

// ---------------------------------------------------------------------------
// Kernel 1: furthest point sampling. One block per batch. Points + running
// min-distances live in registers (16 per thread); LDS argmax tree with
// first-index tie-break to match jnp.argmax.
// ---------------------------------------------------------------------------
__global__ __launch_bounds__(256) void fps_kernel(
    const float* __restrict__ xyz, int* __restrict__ fps_idx) {
  const int b = blockIdx.x, tid = threadIdx.x;
  const float* base = xyz + (size_t)b * N_ * 3;
  float px[16], py[16], pz[16], dd[16];
#pragma unroll
  for (int j = 0; j < 16; j++) {
    int i = tid + j * 256;
    px[j] = base[i * 3 + 0]; py[j] = base[i * 3 + 1]; pz[j] = base[i * 3 + 2];
    dd[j] = 1e10f;
  }
  __shared__ float bc[3];
  __shared__ float rmax[256];
  __shared__ int   ridx[256];
  int last = 0;
  if (tid == 0) fps_idx[b * S_] = 0;
  for (int s = 1; s < S_; s++) {
    if (tid == (last & 255)) {
      int j = last >> 8;
      float sx = px[0], sy = py[0], sz = pz[0];
#pragma unroll
      for (int q = 1; q < 16; q++)
        if (j == q) { sx = px[q]; sy = py[q]; sz = pz[q]; }
      bc[0] = sx; bc[1] = sy; bc[2] = sz;
    }
    __syncthreads();
    float cx = bc[0], cy = bc[1], cz = bc[2];
    float best = -1.0f; int bidx = 0;
#pragma unroll
    for (int j = 0; j < 16; j++) {
      float dx = px[j] - cx, dy = py[j] - cy, dz = pz[j] - cz;
      float d = dx * dx + dy * dy + dz * dz;
      float nd = fminf(dd[j], d);
      dd[j] = nd;
      if (nd > best) { best = nd; bidx = tid + j * 256; }
    }
    rmax[tid] = best; ridx[tid] = bidx;
    __syncthreads();
    for (int off = 128; off > 0; off >>= 1) {
      if (tid < off) {
        float o = rmax[tid + off]; int oi = ridx[tid + off];
        if (o > rmax[tid] || (o == rmax[tid] && oi < ridx[tid])) {
          rmax[tid] = o; ridx[tid] = oi;
        }
      }
      __syncthreads();
    }
    last = ridx[0];
    if (tid == 0) fps_idx[b * S_ + s] = last;
    __syncthreads();
  }
}

// ---------------------------------------------------------------------------
// Kernel 2: KNN (K=32 nearest). One block per centroid; distances in
// registers, 32 serial min-reductions, lower-index tie-break (stable top_k).
// Also writes cen_xyz to output.
// ---------------------------------------------------------------------------
__global__ __launch_bounds__(256) void knn_kernel(
    const float* __restrict__ xyz, const int* __restrict__ fps_idx,
    int* __restrict__ knn_idx, float* __restrict__ cen_out) {
  const int pid = blockIdx.x;          // b*S + s
  const int b = pid >> 10, tid = threadIdx.x;
  const float* base = xyz + (size_t)b * N_ * 3;
  int ci = fps_idx[pid];
  float cx = base[ci * 3 + 0], cy = base[ci * 3 + 1], cz = base[ci * 3 + 2];
  if (tid < 3) cen_out[(size_t)pid * 3 + tid] = base[ci * 3 + tid];
  float dd[16];
#pragma unroll
  for (int j = 0; j < 16; j++) {
    int i = tid + j * 256;
    float dx = base[i * 3] - cx, dy = base[i * 3 + 1] - cy, dz = base[i * 3 + 2] - cz;
    dd[j] = dx * dx + dy * dy + dz * dz;
  }
  __shared__ float rmin[256];
  __shared__ int   ridx[256];
  for (int kk = 0; kk < K_; kk++) {
    float best = 3.0e38f; int bidx = 1 << 30;
#pragma unroll
    for (int j = 0; j < 16; j++)
      if (dd[j] < best) { best = dd[j]; bidx = tid + j * 256; }
    rmin[tid] = best; ridx[tid] = bidx;
    __syncthreads();
    for (int off = 128; off > 0; off >>= 1) {
      if (tid < off) {
        float o = rmin[tid + off]; int oi = ridx[tid + off];
        if (o < rmin[tid] || (o == rmin[tid] && oi < ridx[tid])) {
          rmin[tid] = o; ridx[tid] = oi;
        }
      }
      __syncthreads();
    }
    int widx = ridx[0];
    if (tid == 0) knn_idx[(size_t)pid * K_ + kk] = widx;
    if ((widx & 255) == tid) {
      int j = widx >> 8;
#pragma unroll
      for (int q = 0; q < 16; q++)
        if (j == q) dd[q] = 3.4e38f;
    }
    __syncthreads();
  }
}

// ---------------------------------------------------------------------------
// Kernel 3: per-patch group build + conv stack (GEMM0 128->128, GEMM1
// 128->256, BN+ReLU folded) + max/mean pool over K in fragments.
// One block (8 waves) per patch.
// ---------------------------------------------------------------------------
__global__ __launch_bounds__(256) void patch_kernel(
    const float* __restrict__ feature, const int* __restrict__ fps_idx,
    const int* __restrict__ knn_idx,
    const _Float16* __restrict__ w0h, const _Float16* __restrict__ w1h,
    const float* __restrict__ sc0, const float* __restrict__ bi0,
    const float* __restrict__ sc1, const float* __restrict__ bi1,
    float* __restrict__ maxp_ws, float* __restrict__ avg_out) {
  const int pid = blockIdx.x;          // b*S + s
  const int b = pid >> 10;
  const int tid = threadIdx.x, wave = tid >> 5, lane = tid & 31;

  __shared__ __align__(16) _Float16 grp[32 * 136];  // rows padded: 272B stride
  __shared__ __align__(16) _Float16 h1[32 * 136];
  __shared__ float cenf[C_];
  __shared__ int   knnS[K_];

  const float* fb = feature + (size_t)b * N_ * C_;
  int ci = fps_idx[pid];
  if (tid < C_) cenf[tid] = fb[(size_t)ci * C_ + tid];
  if (tid < K_) knnS[tid] = knn_idx[(size_t)pid * K_ + tid];
  __syncthreads();

  // groups[k][c]: c<64 -> centroid feature; else neighbor - centroid
  for (int e = tid; e < 32 * 128; e += 256) {
    int k = e >> 7, c = e & 127;
    float v = (c < C_) ? cenf[c]
                       : (fb[(size_t)knnS[k] * C_ + (c - C_)] - cenf[c - C_]);
    grp[k * 136 + c] = (_Float16)v;
  }
  __syncthreads();

  // GEMM0: [32x128] x [128x128]^T ; wave w owns N-tile w
  {
    const int nt = wave;
    v8f c0 = {}, c1 = {};
#pragma unroll
    for (int kc = 0; kc < 4; kc++) {
      v16h bfr = load_frag(w0h + nt * 16 * 128 + kc * 32, 128);
      v16h a0 = load_frag(grp + kc * 32, 136);
      v16h a1 = load_frag(grp + 16 * 136 + kc * 32, 136);
      c0 = wmma_f16(a0, bfr, c0);
      c1 = wmma_f16(a1, bfr, c1);
    }
    int n = nt * 16 + (lane & 15);
    float sA = sc0[n], sB = bi0[n];
    int mh = (lane >> 4) * 8;
#pragma unroll
    for (int r = 0; r < 8; r++) {
      h1[(r + mh) * 136 + n]      = (_Float16)fmaxf(c0[r] * sA + sB, 0.0f);
      h1[(r + mh + 16) * 136 + n] = (_Float16)fmaxf(c1[r] * sA + sB, 0.0f);
    }
  }
  __syncthreads();

  // GEMM1: [32x128] x [128x256]^T ; wave w owns N-tiles 2w, 2w+1.
  // Pool over M (the K=32 patch points) directly from fragments.
#pragma unroll
  for (int t = 0; t < 2; t++) {
    const int nt = wave * 2 + t;
    v8f c0 = {}, c1 = {};
#pragma unroll
    for (int kc = 0; kc < 4; kc++) {
      v16h bfr = load_frag(w1h + nt * 16 * 128 + kc * 32, 128);
      v16h a0 = load_frag(h1 + kc * 32, 136);
      v16h a1 = load_frag(h1 + 16 * 136 + kc * 32, 136);
      c0 = wmma_f16(a0, bfr, c0);
      c1 = wmma_f16(a1, bfr, c1);
    }
    int n = nt * 16 + (lane & 15);
    float sA = sc1[n], sB = bi1[n];
    float mx = 0.0f, sm = 0.0f;   // relu outputs are >= 0
#pragma unroll
    for (int r = 0; r < 8; r++) {
      float v0 = fmaxf(c0[r] * sA + sB, 0.0f);
      float v1 = fmaxf(c1[r] * sA + sB, 0.0f);
      mx = fmaxf(mx, fmaxf(v0, v1));
      sm += v0 + v1;
    }
    mx = fmaxf(mx, __shfl_xor(mx, 16, 32));
    sm += __shfl_xor(sm, 16, 32);
    if (lane < 16) {
      size_t o = (size_t)pid * D1_ + nt * 16 + lane;
      maxp_ws[o] = mx;
      avg_out[o] = sm * (1.0f / 32.0f);
    }
  }
}

// ---------------------------------------------------------------------------
// Kernel 4: ResMLP over maxp [8192 x 256]: relu(bn(x*rw1)) -> bn(.*rw2),
// residual add + relu. 16 rows per block.
// ---------------------------------------------------------------------------
__global__ __launch_bounds__(256) void resmlp_kernel(
    const float* __restrict__ maxp_ws,
    const _Float16* __restrict__ rw1h, const _Float16* __restrict__ rw2h,
    const float* __restrict__ scr1, const float* __restrict__ bir1,
    const float* __restrict__ scr2, const float* __restrict__ bir2,
    float* __restrict__ maxp_out) {
  const int row0 = blockIdx.x * 16;
  const int tid = threadIdx.x, wave = tid >> 5, lane = tid & 31;
  __shared__ __align__(16) _Float16 xa[16 * 264];   // 256 + 8 pad halves
  __shared__ __align__(16) _Float16 h1r[16 * 264];

  for (int e = tid; e < 16 * 256; e += 256) {
    int r = e >> 8, c = e & 255;
    xa[r * 264 + c] = (_Float16)maxp_ws[(size_t)(row0 + r) * D1_ + c];
  }
  __syncthreads();

#pragma unroll
  for (int t = 0; t < 2; t++) {
    const int nt = wave * 2 + t;
    v8f acc = {};
#pragma unroll
    for (int kc = 0; kc < 8; kc++) {
      v16h a = load_frag(xa + kc * 32, 264);
      v16h bf = load_frag(rw1h + nt * 16 * 256 + kc * 32, 256);
      acc = wmma_f16(a, bf, acc);
    }
    int n = nt * 16 + (lane & 15);
    float sA = scr1[n], sB = bir1[n];
    int mh = (lane >> 4) * 8;
#pragma unroll
    for (int r = 0; r < 8; r++)
      h1r[(r + mh) * 264 + n] = (_Float16)fmaxf(acc[r] * sA + sB, 0.0f);
  }
  __syncthreads();

#pragma unroll
  for (int t = 0; t < 2; t++) {
    const int nt = wave * 2 + t;
    v8f acc = {};
#pragma unroll
    for (int kc = 0; kc < 8; kc++) {
      v16h a = load_frag(h1r + kc * 32, 264);
      v16h bf = load_frag(rw2h + nt * 16 * 256 + kc * 32, 256);
      acc = wmma_f16(a, bf, acc);
    }
    int n = nt * 16 + (lane & 15);
    float sA = scr2[n], sB = bir2[n];
    int mh = (lane >> 4) * 8;
#pragma unroll
    for (int r = 0; r < 8; r++) {
      size_t idx = (size_t)(row0 + r + mh) * D1_ + n;
      float v = acc[r] * sA + sB + maxp_ws[idx];   // residual
      maxp_out[idx] = fmaxf(v, 0.0f);
    }
  }
}

// ---------------------------------------------------------------------------
extern "C" void kernel_launch(void* const* d_in, const int* in_sizes, int n_in,
                              void* d_out, int out_size, void* d_ws, size_t ws_size,
                              hipStream_t stream) {
  const float* xyz     = (const float*)d_in[0];
  const float* feature = (const float*)d_in[1];
  const float* w0  = (const float*)d_in[2];
  const float* b0  = (const float*)d_in[3];
  const float* g0  = (const float*)d_in[4];
  const float* be0 = (const float*)d_in[5];
  const float* m0  = (const float*)d_in[6];
  const float* v0  = (const float*)d_in[7];
  const float* w1  = (const float*)d_in[8];
  const float* b1  = (const float*)d_in[9];
  const float* g1  = (const float*)d_in[10];
  const float* be1 = (const float*)d_in[11];
  const float* m1  = (const float*)d_in[12];
  const float* v1  = (const float*)d_in[13];
  const float* rw1  = (const float*)d_in[14];
  const float* rb1  = (const float*)d_in[15];
  const float* rg1  = (const float*)d_in[16];
  const float* rbe1 = (const float*)d_in[17];
  const float* rm1  = (const float*)d_in[18];
  const float* rv1  = (const float*)d_in[19];
  const float* rw2  = (const float*)d_in[20];
  const float* rb2  = (const float*)d_in[21];
  const float* rg2  = (const float*)d_in[22];
  const float* rbe2 = (const float*)d_in[23];
  const float* rm2  = (const float*)d_in[24];
  const float* rv2  = (const float*)d_in[25];

  // workspace carve-out (256B aligned slices)
  char* ws = (char*)d_ws;
  size_t off = 0;
  auto alloc = [&](size_t bytes) -> void* {
    void* p = ws + off;
    off = (off + bytes + 255) & ~(size_t)255;
    return p;
  };
  int*      fps_idx = (int*)alloc((size_t)B_ * S_ * 4);
  int*      knn_idx = (int*)alloc((size_t)B_ * S_ * K_ * 4);
  float*    maxp_ws = (float*)alloc((size_t)B_ * S_ * D1_ * 4);
  _Float16* w0h  = (_Float16*)alloc((size_t)D0_ * 2 * C_ * 2);
  _Float16* w1h  = (_Float16*)alloc((size_t)D1_ * D0_ * 2);
  _Float16* rw1h = (_Float16*)alloc((size_t)D1_ * D1_ * 2);
  _Float16* rw2h = (_Float16*)alloc((size_t)D1_ * D1_ * 2);
  float* sc0  = (float*)alloc(D0_ * 4);
  float* bi0  = (float*)alloc(D0_ * 4);
  float* sc1  = (float*)alloc(D1_ * 4);
  float* bi1  = (float*)alloc(D1_ * 4);
  float* scr1 = (float*)alloc(D1_ * 4);
  float* bir1 = (float*)alloc(D1_ * 4);
  float* scr2 = (float*)alloc(D1_ * 4);
  float* bir2 = (float*)alloc(D1_ * 4);

  // output layout: cen_xyz [B,S,3] | maxp [B,S,256] | avgp [B,S,256]
  float* cen_out  = (float*)d_out;
  float* maxp_out = cen_out + (size_t)B_ * S_ * 3;
  float* avg_out  = maxp_out + (size_t)B_ * S_ * D1_;

  prep_kernel<<<256, 256, 0, stream>>>(
      w0, w1, rw1, rw2,
      b0, g0, be0, m0, v0,
      b1, g1, be1, m1, v1,
      rb1, rg1, rbe1, rm1, rv1,
      rb2, rg2, rbe2, rm2, rv2,
      w0h, w1h, rw1h, rw2h,
      sc0, bi0, sc1, bi1, scr1, bir1, scr2, bir2);

  fps_kernel<<<B_, 256, 0, stream>>>(xyz, fps_idx);

  knn_kernel<<<B_ * S_, 256, 0, stream>>>(xyz, fps_idx, knn_idx, cen_out);

  patch_kernel<<<B_ * S_, 256, 0, stream>>>(
      feature, fps_idx, knn_idx, w0h, w1h, sc0, bi0, sc1, bi1,
      maxp_ws, avg_out);

  resmlp_kernel<<<(B_ * S_) / 16, 256, 0, stream>>>(
      maxp_ws, rw1h, rw2h, scr1, bir1, scr2, bir2, maxp_out);
}
